// RelationNetworks_78520592105985
// MI455X (gfx1250) — matrix operand order repaired and checked by reading, not compiled, not run on hardware
//
#include <hip/hip_runtime.h>
#include <hip/hip_bf16.h>
#include <stdint.h>

typedef __attribute__((ext_vector_type(16))) __bf16 v16bf;
typedef __attribute__((ext_vector_type(8)))  float  v8f;
typedef __attribute__((ext_vector_type(4)))  unsigned int u32x4;
typedef __attribute__((ext_vector_type(8)))  int i32x8;
typedef __attribute__((ext_vector_type(4)))  int i32x4;

#if __has_builtin(__builtin_amdgcn_tensor_load_to_lds) && __has_builtin(__builtin_amdgcn_s_wait_tensorcnt)
#define HAVE_TDM 1
#else
#define HAVE_TDM 0
#endif

#define CH 24
#define NC 26
#define EH 64
#define LH 128
#define MH 256
#define LQ 16
#define BATCH 4
#define CLASSES 3129
#define NPAIR 196
#define NROWS (NPAIR*NPAIR)        /* 38416 rows per batch item */
#define TM 96                      /* rows per block (6 waves x 16 rows) */
#define NWAVES 6
#define THREADS (NWAVES*32)        /* 192 */
#define NTILES ((NROWS + TM - 1) / TM)   /* 401 */
#define LDS_STRIDE 264             /* bf16 row stride: 16B-aligned, bank-spread */

/* LDS byte layout of the fused g-kernel */
#define OFF_A    0
#define OFF_W0   (TM * LDS_STRIDE * 2)                 /*  50,688 */
#define OFF_W1   (OFF_W0 + MH * LDS_STRIDE * 2)        /* 185,856 */
#define OFF_SUMS (OFF_W1 + MH * LDS_STRIDE * 2)        /* 321,024 */
#define LDS_BYTES (OFF_SUMS + 256 * 4)                 /* 322,048 <= 320KB WGP LDS */

/* ---------------- helpers ---------------- */

__device__ inline __bf16 f2bf(float f) {
  union { float f; uint32_t u; } c; c.f = f;
  uint32_t r = c.u + 0x7FFFu + ((c.u >> 16) & 1u);   /* RNE */
  union { uint16_t h; __bf16 b; } o; o.h = (uint16_t)(r >> 16);
  return o.b;
}

union Frag16 { uint4 q[2]; v16bf v; };

/* A fragment: 16x32 bf16, rows rowBase..+15 (row-major, stride LDS_STRIDE). */
__device__ inline v16bf load_a(const __bf16* Abase, int rowBase, int k0, int lane) {
  int m = lane & 15, kg = lane >> 4;
  const __bf16* p = Abase + (rowBase + m) * LDS_STRIDE + k0 + kg * 8;
  Frag16 f;
  f.q[0] = *(const uint4*)p;
  f.q[1] = *(const uint4*)(p + 16);
  return f.v;
}

/* B fragment: 32x16 of W[k][n] stored transposed in LDS as Wt[n][k]. */
__device__ inline v16bf load_b(const __bf16* Wt, int colBase, int k0, int lane) {
  int n = lane & 15, kg = lane >> 4;
  const __bf16* p = Wt + (colBase + n) * LDS_STRIDE + k0 + kg * 8;
  Frag16 f;
  f.q[0] = *(const uint4*)p;
  f.q[1] = *(const uint4*)(p + 16);
  return f.v;
}

#if HAVE_TDM
/* TDM: DMA one 256x128-dword weight tile (bf16 256x256) from global into LDS,
   inserting 4 pad dwords after every 128 dwords -> LDS_STRIDE=264 bf16 rows.
   D# per CDNA5 ISA. Tracked by TENSORcnt.
   6-arg builtin form (confirmed by this toolchain's diagnostic). */
__device__ inline void tdm_load_w(const __bf16* src, uint32_t ldsOff) {
  uint64_t ga = (uint64_t)(uintptr_t)src;
  u32x4 g0;
  g0[0] = 1u;                                   /* count=1, user descriptor */
  g0[1] = ldsOff;                               /* lds_addr (bytes) */
  g0[2] = (uint32_t)ga;                         /* global_addr[31:0] */
  g0[3] = (uint32_t)((ga >> 32) & 0x1FFFFFFu)   /* global_addr[56:32] */
        | (2u << 30);                           /* type=2 ("image") */
  i32x8 g1;
  g1[0] = (int)((2u << 16)      /* data_size = 4B (dword) */
              | (1u << 20)      /* pad_enable */
              | (6u << 22)      /* pad_interval enc 6 = 128 dwords */
              | (3u << 25));    /* pad_amount  enc 3 = 4 dwords */
  g1[1] = (int)(128u << 16);    /* tensor_dim0 = 128 dwords/row (bits 79:48) */
  g1[2] = (int)(256u << 16);    /* tensor_dim1 = 256 rows (bits 111:80) */
  g1[3] = (int)(128u << 16);    /* tile_dim0 = 128 (bits 127:112) */
  g1[4] = (int)256;             /* tile_dim1 = 256 (bits 143:128) */
  g1[5] = (int)128;             /* tensor_dim0_stride = 128 dwords (bits 207:160) */
  g1[6] = 0;
  g1[7] = 0;
  i32x4 z4 = {0, 0, 0, 0};
  i32x8 z8 = {0, 0, 0, 0, 0, 0, 0, 0};
  __builtin_amdgcn_tensor_load_to_lds(g0, g1, z4, z4, z8, 0);
}
#endif

/* ---------------- conv + batchnorm ---------------- */

__global__ void conv3x3s2(const float* __restrict__ in, const float* __restrict__ w,
                          float* __restrict__ out, int Bn, int Cin, int Cout,
                          int Hin, int Win, int Hout, int Wout) {
  int idx = blockIdx.x * blockDim.x + threadIdx.x;
  int total = Bn * Cout * Hout * Wout;
  if (idx >= total) return;
  int ow = idx % Wout; int t = idx / Wout;
  int oh = t % Hout;   t /= Hout;
  int oc = t % Cout;   int bb = t / Cout;
  int ih0 = oh * 2 - 1, iw0 = ow * 2 - 1;
  float acc = 0.f;
  for (int ic = 0; ic < Cin; ++ic) {
    const float* ip = in + (size_t)((bb * Cin + ic) * Hin) * Win;
    const float* wp = w + ((oc * Cin + ic) * 3) * 3;
    #pragma unroll
    for (int kh = 0; kh < 3; ++kh) {
      int ih = ih0 + kh;
      if ((unsigned)ih >= (unsigned)Hin) continue;
      #pragma unroll
      for (int kw = 0; kw < 3; ++kw) {
        int iw = iw0 + kw;
        if ((unsigned)iw >= (unsigned)Win) continue;
        acc += ip[ih * Win + iw] * wp[kh * 3 + kw];
      }
    }
  }
  out[idx] = acc;
}

__global__ __launch_bounds__(256)
void bn_stats(const float* __restrict__ x, const float* __restrict__ g,
              const float* __restrict__ bta, float* __restrict__ stats,
              int Bn, int C, int HW) {
  __shared__ float s1[256], s2[256];
  int c = blockIdx.x, tid = threadIdx.x;
  float sum = 0.f, sq = 0.f;
  for (int b2 = 0; b2 < Bn; ++b2) {
    const float* p = x + (size_t)(b2 * C + c) * HW;
    for (int i = tid; i < HW; i += 256) { float v = p[i]; sum += v; sq += v * v; }
  }
  s1[tid] = sum; s2[tid] = sq; __syncthreads();
  for (int st = 128; st > 0; st >>= 1) {
    if (tid < st) { s1[tid] += s1[tid + st]; s2[tid] += s2[tid + st]; }
    __syncthreads();
  }
  if (tid == 0) {
    float n = (float)(Bn * HW);
    float m = s1[0] / n;
    float var = s2[0] / n - m * m;
    float sc = g[c] * rsqrtf(var + 1e-5f);
    stats[c] = sc;
    stats[C + c] = bta[c] - m * sc;
  }
}

__global__ void bn_apply(float* __restrict__ x, const float* __restrict__ stats,
                         int C, int HW, int total) {
  int idx = blockIdx.x * blockDim.x + threadIdx.x;
  if (idx >= total) return;
  int c = (idx / HW) % C;
  x[idx] = fmaxf(x[idx] * stats[c] + stats[C + c], 0.f);
}

/* ---------------- LSTM (single block, 4 batches x 128 units) ---------------- */

__global__ __launch_bounds__(512)
void lstm_kernel(const int* __restrict__ q, const int* __restrict__ qlen,
                 const float* __restrict__ embW, const float* __restrict__ Wih,
                 const float* __restrict__ Whh, const float* __restrict__ bih,
                 const float* __restrict__ bhh, float* __restrict__ hT) {
  __shared__ float hs[BATCH][LH], cs[BATCH][LH];
  int tid = threadIdx.x;
  int bb = tid >> 7, u = tid & 127;
  hs[bb][u] = 0.f; cs[bb][u] = 0.f;
  __syncthreads();
  for (int t = 0; t < LQ; ++t) {
    int qi = q[bb * LQ + t];
    qi = qi < 0 ? 0 : (qi > 63 ? 63 : qi);
    const float* x = embW + qi * EH;
    float gate[4];
    #pragma unroll
    for (int gi = 0; gi < 4; ++gi) {
      int row = gi * LH + u;
      float acc = bih[row] + bhh[row];
      const float* wi = Wih + row * EH;
      for (int e = 0; e < EH; ++e) acc += x[e] * wi[e];
      const float* wh = Whh + row * LH;
      for (int e = 0; e < LH; ++e) acc += hs[bb][e] * wh[e];
      gate[gi] = acc;
    }
    float ig = 1.f / (1.f + expf(-gate[0]));
    float fg = 1.f / (1.f + expf(-gate[1]));
    float gg = tanhf(gate[2]);
    float og = 1.f / (1.f + expf(-gate[3]));
    float cn = fg * cs[bb][u] + ig * gg;
    float hn = og * tanhf(cn);
    __syncthreads();
    if (t < qlen[bb]) { hs[bb][u] = hn; cs[bb][u] = cn; }
    __syncthreads();
  }
  hT[bb * LH + u] = hs[bb][u];
}

/* ---------------- p1 / p2 / ph projections ---------------- */

__global__ void p1p2_kernel(const float* __restrict__ x4, const float* __restrict__ gW1,
                            float* __restrict__ p1, float* __restrict__ p2) {
  int idx = blockIdx.x * blockDim.x + threadIdx.x;
  if (idx >= BATCH * NPAIR * MH) return;
  int n = idx & 255; int t = idx >> 8;
  int p = t % NPAIR; int bb = t / NPAIR;
  int hh = p / 14, wwi = p % 14;
  float ct[NC];
  #pragma unroll
  for (int c2 = 0; c2 < CH; ++c2) ct[c2] = x4[((bb * CH + c2) * 14 + hh) * 14 + wwi];
  ct[24] = -4.f + 8.f * (float)wwi / 13.f;   /* xx channel */
  ct[25] = -4.f + 8.f * (float)hh  / 13.f;   /* yy channel */
  float a1 = 0.f, a2 = 0.f;
  #pragma unroll
  for (int c2 = 0; c2 < NC; ++c2) {
    a1 += ct[c2] * gW1[c2 * MH + n];
    a2 += ct[c2] * gW1[(NC + c2) * MH + n];
  }
  p1[idx] = a1;
  p2[idx] = a2;
}

__global__ void ph_kernel(const float* __restrict__ hT, const float* __restrict__ gW1,
                          const float* __restrict__ gb1, float* __restrict__ ph) {
  int idx = blockIdx.x * blockDim.x + threadIdx.x;
  if (idx >= BATCH * MH) return;
  int n = idx & 255; int bb = idx >> 8;
  float acc = gb1[n];
  for (int u = 0; u < LH; ++u) acc += hT[bb * LH + u] * gW1[(2 * NC + u) * MH + n];
  ph[idx] = acc;
}

/* ------- weight convert: f32 row-major [k][n] -> bf16 transposed [n][k] ------ */

__global__ void wt_convert(const float* __restrict__ src, __bf16* __restrict__ dst) {
  int idx = blockIdx.x * blockDim.x + threadIdx.x;
  if (idx >= MH * MH) return;
  int k = idx >> 8, n = idx & 255;
  dst[n * MH + k] = f2bf(src[idx]);
}

__global__ void zero_f32(float* __restrict__ p, int n) {
  int i = blockIdx.x * blockDim.x + threadIdx.x;
  if (i < n) p[i] = 0.f;
}

/* -------- fused 3-layer g-MLP: WMMA + TDM weight prefetch + reduction ------- */

__global__ __launch_bounds__(THREADS)
void rn_g_kernel(const float* __restrict__ p1, const float* __restrict__ p2,
                 const float* __restrict__ ph, const __bf16* __restrict__ wt,
                 const float* __restrict__ gb2, const float* __restrict__ gb3,
                 const float* __restrict__ gb4, float* __restrict__ gsum) {
  extern __shared__ __align__(16) char smem[];
  __bf16* A0 = (__bf16*)(smem + OFF_A);           /* TM x LDS_STRIDE, in-place */
  __bf16* Wb0 = (__bf16*)(smem + OFF_W0);
  __bf16* Wb1 = (__bf16*)(smem + OFF_W1);
  float* sums = (float*)(smem + OFF_SUMS);

  const int tid = threadIdx.x;
  const int wave = tid >> 5;
  const int lane = tid & 31;

  const int b = blockIdx.x / NTILES;
  const int tile = blockIdx.x % NTILES;
  const int row0 = tile * TM;
  int nvalid = NROWS - row0; if (nvalid > TM) nvalid = TM;

  for (int i = tid; i < 256; i += THREADS) sums[i] = 0.f;

#if HAVE_TDM
  /* kick off layer-0 weight DMA; overlaps with a0 construction below */
  if (tid < 32) tdm_load_w(wt, OFF_W0);
#endif

  /* build a0 = relu(p1[i] + p2[j] + ph) as bf16 in LDS */
  const float* p1b = p1 + (size_t)b * NPAIR * MH;
  const float* p2b = p2 + (size_t)b * NPAIR * MH;
  const float* phb = ph + b * MH;
  for (int idx = tid; idx < TM * MH; idx += THREADS) {
    int r = idx >> 8;
    int c = idx & 255;
    float v = 0.f;
    int gr = row0 + r;
    if (gr < NROWS) {
      int i = gr % NPAIR;
      int j = gr / NPAIR;
      v = fmaxf(p1b[i * MH + c] + p2b[j * MH + c] + phb[c], 0.f);
    }
    A0[r * LDS_STRIDE + c] = f2bf(v);
  }

  const int rowBase = wave * 16;
  const int n = lane & 15, kg = lane >> 4;
  const v8f vzero = {0.f, 0.f, 0.f, 0.f, 0.f, 0.f, 0.f, 0.f};

  for (int layer = 0; layer < 3; ++layer) {
    const __bf16* WL = (layer & 1) ? Wb1 : Wb0;
#if HAVE_TDM
    if (tid < 32) __builtin_amdgcn_s_wait_tensorcnt(0);  /* W[layer&1] landed */
    __syncthreads();   /* all waves: weights visible; prior layer's reads done */
    /* prefetch next layer's weights into the other buffer while we compute */
    if (layer < 2 && tid < 32)
      tdm_load_w(wt + (size_t)(layer + 1) * MH * MH, (layer & 1) ? OFF_W0 : OFF_W1);
#else
    __syncthreads();
    {
      const __bf16* wsrc = wt + (size_t)layer * MH * MH;
      __bf16* wdst = (layer & 1) ? Wb1 : Wb0;
      for (int idx = tid; idx < MH * MH / 2; idx += THREADS) {
        int nn = idx >> 7;
        int kd = idx & 127;
        ((uint32_t*)(wdst + nn * LDS_STRIDE))[kd] = ((const uint32_t*)(wsrc + nn * MH))[kd];
      }
    }
    __syncthreads();
#endif
    const float* bias = (layer == 0) ? gb2 : ((layer == 1) ? gb3 : gb4);

    v8f acc[16];
    #pragma unroll
    for (int t = 0; t < 16; ++t) acc[t] = vzero;

    for (int kk = 0; kk < MH; kk += 32) {
      v16bf a = load_a(A0, rowBase, kk, lane);
      v16bf bcur = load_b(WL, 0, kk, lane);
      #pragma unroll
      for (int t = 0; t < 16; ++t) {
        /* issue next B-fragment load before consuming the current one */
        v16bf bnext = load_b(WL, ((t + 1) & 15) * 16, kk, lane);
        acc[t] = __builtin_amdgcn_wmma_f32_16x16x32_bf16(
            false, a, false, bcur, (short)0, acc[t], false, false);
        bcur = bnext;
      }
    }

    if (layer < 2) {
      /* bias + relu, write activation IN PLACE: each wave owns its 16 rows and
         has finished reading all their K before any write. */
      #pragma unroll
      for (int t = 0; t < 16; ++t) {
        int col = t * 16 + n;
        float bv = bias[col];
        #pragma unroll
        for (int r = 0; r < 8; ++r) {
          int m = r + 8 * kg;
          float v = fmaxf(acc[t][r] + bv, 0.f);
          A0[(rowBase + m) * LDS_STRIDE + col] = f2bf(v);
        }
      }
    } else {
      /* final layer: bias + relu + masked row-sum into LDS accumulator */
      #pragma unroll
      for (int t = 0; t < 16; ++t) {
        int col = t * 16 + n;
        float bv = bias[col];
        float s = 0.f;
        #pragma unroll
        for (int r = 0; r < 8; ++r) {
          int m = r + 8 * kg;
          float v = fmaxf(acc[t][r] + bv, 0.f);
          if (rowBase + m < nvalid) s += v;
        }
        atomicAdd(&sums[col], s);
      }
    }
  }
  __syncthreads();
  for (int i = tid; i < 256; i += THREADS) atomicAdd(&gsum[b * MH + i], sums[i]);
}

/* ---------------- f-MLP + classifier (one block per batch item) ------------ */

__global__ __launch_bounds__(256)
void fmlp_kernel(const float* __restrict__ gsum,
                 const float* __restrict__ fW1, const float* __restrict__ fb1,
                 const float* __restrict__ fW2, const float* __restrict__ fb2,
                 const float* __restrict__ fW3, const float* __restrict__ fb3,
                 float* __restrict__ out) {
  __shared__ float z1[MH], z2[MH];
  int bb = blockIdx.x, tid = threadIdx.x;
  const float* gs = gsum + bb * MH;
  float acc = fb1[tid];
  for (int k = 0; k < MH; ++k) acc += gs[k] * fW1[k * MH + tid];
  z1[tid] = fmaxf(acc, 0.f);
  __syncthreads();
  acc = fb2[tid];
  for (int k = 0; k < MH; ++k) acc += z1[k] * fW2[k * MH + tid];
  z2[tid] = fmaxf(acc, 0.f);
  __syncthreads();
  for (int nn = tid; nn < CLASSES; nn += 256) {
    float o = fb3[nn];
    for (int k = 0; k < MH; ++k) o += z2[k] * fW3[k * CLASSES + nn];
    out[bb * CLASSES + nn] = o;
  }
}

/* ---------------- host launch ---------------- */

extern "C" void kernel_launch(void* const* d_in, const int* in_sizes, int n_in,
                              void* d_out, int out_size, void* d_ws, size_t ws_size,
                              hipStream_t stream) {
  (void)in_sizes; (void)n_in; (void)out_size; (void)ws_size;

  const float* image   = (const float*)d_in[0];
  const int*   question= (const int*)  d_in[1];
  const int*   qlen    = (const int*)  d_in[2];
  const float* k1 = (const float*)d_in[3];
  const float* g1 = (const float*)d_in[4];
  const float* b1 = (const float*)d_in[5];
  const float* k2 = (const float*)d_in[6];
  const float* g2 = (const float*)d_in[7];
  const float* b2 = (const float*)d_in[8];
  const float* k3 = (const float*)d_in[9];
  const float* g3 = (const float*)d_in[10];
  const float* b3 = (const float*)d_in[11];
  const float* k4 = (const float*)d_in[12];
  const float* g4 = (const float*)d_in[13];
  const float* b4 = (const float*)d_in[14];
  const float* embW = (const float*)d_in[15];
  const float* Wih  = (const float*)d_in[16];
  const float* Whh  = (const float*)d_in[17];
  const float* bih  = (const float*)d_in[18];
  const float* bhh  = (const float*)d_in[19];
  const float* gW1  = (const float*)d_in[20];
  const float* gb1  = (const float*)d_in[21];
  const float* gW2  = (const float*)d_in[22];
  const float* gb2  = (const float*)d_in[23];
  const float* gW3  = (const float*)d_in[24];
  const float* gb3  = (const float*)d_in[25];
  const float* gW4  = (const float*)d_in[26];
  const float* gb4  = (const float*)d_in[27];
  const float* fW1  = (const float*)d_in[28];
  const float* fb1  = (const float*)d_in[29];
  const float* fW2  = (const float*)d_in[30];
  const float* fb2  = (const float*)d_in[31];
  const float* fW3  = (const float*)d_in[32];
  const float* fb3  = (const float*)d_in[33];
  float* out = (float*)d_out;

  /* workspace layout (bytes) */
  char* ws = (char*)d_ws;
  float*  bufA  = (float*)(ws + 0);          /* 4*24*112*112 f32 = 4,816,896 B */
  float*  bufB  = (float*)(ws + 4816896);    /* 4*24*56*56  f32 = 1,204,224 B */
  float*  hT    = (float*)(ws + 6021120);    /* 4*128 f32 */
  float*  phb   = (float*)(ws + 6023168);    /* 4*256 f32 */
  __bf16* wtb   = (__bf16*)(ws + 6027264);   /* 3*256*256 bf16 = 393,216 B */
  float*  gsum  = (float*)(ws + 6420480);    /* 4*256 f32 */
  float*  stats = (float*)(ws + 6424576);    /* 2*24 f32, reused per BN layer */
  float*  p1 = bufA;                         /* reuses conv scratch (x3 dead) */
  float*  p2 = bufA + BATCH * NPAIR * MH;

  const int TB = 256;
  auto cdiv = [](int a, int b) { return (a + b - 1) / b; };

  /* conv1: (4,3,224,224) -> (4,24,112,112) in bufA */
  {
    int tot = BATCH * CH * 112 * 112;
    conv3x3s2<<<cdiv(tot, TB), TB, 0, stream>>>(image, k1, bufA, BATCH, 3, CH, 224, 224, 112, 112);
    bn_stats<<<CH, 256, 0, stream>>>(bufA, g1, b1, stats, BATCH, CH, 112 * 112);
    bn_apply<<<cdiv(tot, TB), TB, 0, stream>>>(bufA, stats, CH, 112 * 112, tot);
  }
  /* conv2 -> bufB (56) */
  {
    int tot = BATCH * CH * 56 * 56;
    conv3x3s2<<<cdiv(tot, TB), TB, 0, stream>>>(bufA, k2, bufB, BATCH, CH, CH, 112, 112, 56, 56);
    bn_stats<<<CH, 256, 0, stream>>>(bufB, g2, b2, stats, BATCH, CH, 56 * 56);
    bn_apply<<<cdiv(tot, TB), TB, 0, stream>>>(bufB, stats, CH, 56 * 56, tot);
  }
  /* conv3 -> bufA (28) */
  {
    int tot = BATCH * CH * 28 * 28;
    conv3x3s2<<<cdiv(tot, TB), TB, 0, stream>>>(bufB, k3, bufA, BATCH, CH, CH, 56, 56, 28, 28);
    bn_stats<<<CH, 256, 0, stream>>>(bufA, g3, b3, stats, BATCH, CH, 28 * 28);
    bn_apply<<<cdiv(tot, TB), TB, 0, stream>>>(bufA, stats, CH, 28 * 28, tot);
  }
  /* conv4 -> bufB (14) = x4 */
  {
    int tot = BATCH * CH * 14 * 14;
    conv3x3s2<<<cdiv(tot, TB), TB, 0, stream>>>(bufA, k4, bufB, BATCH, CH, CH, 28, 28, 14, 14);
    bn_stats<<<CH, 256, 0, stream>>>(bufB, g4, b4, stats, BATCH, CH, 14 * 14);
    bn_apply<<<cdiv(tot, TB), TB, 0, stream>>>(bufB, stats, CH, 14 * 14, tot);
  }

  /* LSTM */
  lstm_kernel<<<1, 512, 0, stream>>>(question, qlen, embW, Wih, Whh, bih, bhh, hT);

  /* bf16 transposed weights for the three g layers */
  wt_convert<<<cdiv(MH * MH, TB), TB, 0, stream>>>(gW2, wtb);
  wt_convert<<<cdiv(MH * MH, TB), TB, 0, stream>>>(gW3, wtb + MH * MH);
  wt_convert<<<cdiv(MH * MH, TB), TB, 0, stream>>>(gW4, wtb + 2 * MH * MH);

  /* projections */
  {
    int tot = BATCH * NPAIR * MH;
    p1p2_kernel<<<cdiv(tot, TB), TB, 0, stream>>>(bufB, gW1, p1, p2);
  }
  ph_kernel<<<cdiv(BATCH * MH, TB), TB, 0, stream>>>(hT, gW1, gb1, phb);

  /* zero gsum every call (determinism; harness does not re-poison) */
  zero_f32<<<cdiv(BATCH * MH, TB), TB, 0, stream>>>(gsum, BATCH * MH);

  /* fused 3-layer WMMA g-MLP with TDM-prefetched double-buffered weights */
  (void)hipFuncSetAttribute((const void*)rn_g_kernel,
                            hipFuncAttributeMaxDynamicSharedMemorySize, LDS_BYTES);
  rn_g_kernel<<<BATCH * NTILES, THREADS, LDS_BYTES, stream>>>(
      p1, p2, phb, wtb, gb2, gb3, gb4, gsum);

  /* f-MLP + classifier */
  fmlp_kernel<<<BATCH, 256, 0, stream>>>(gsum, fW1, fb1, fW2, fb2, fW3, fb3, out);
}
